// AdditiveAttention_19284403159044
// MI455X (gfx1250) — matrix operand order repaired and checked by reading, not compiled, not run on hardware
//
#include <hip/hip_runtime.h>
#include <hip/hip_bf16.h>

typedef __attribute__((ext_vector_type(16))) _Float16 v16h;
typedef __attribute__((ext_vector_type(8)))  _Float16 v8h;
typedef __attribute__((ext_vector_type(8)))  float    v8f;

// Problem constants (fixed-shape workload)
constexpr int Bb = 32;
constexpr int Nn = 128;
constexpr int Dd = 512;   // K of GEMMs
constexpr int Aa = 512;   // hidden dim
constexpr int Mm = Bb * Nn;   // 4096 GEMM rows

// ---------------------------------------------------------------------------
// fast tanh: prefer hardware v_tanh_f32 (gfx1250 trans op) if builtin exists,
// else 2-trans formulation tanh(x) = 1 - 2/(exp2(2x*log2e)+1).
// Both saturate correctly at +/-1 for large |x|.
// ---------------------------------------------------------------------------
__device__ __forceinline__ float fast_tanh(float x) {
#if __has_builtin(__builtin_amdgcn_tanhf)
  return __builtin_amdgcn_tanhf(x);
#else
  float t = __builtin_amdgcn_exp2f(x * 2.885390081777927f); // 2*log2(e)
  float r = __builtin_amdgcn_rcpf(t + 1.0f);
  return 1.0f - 2.0f * r;
#endif
}

// ---------------------------------------------------------------------------
// f32 -> f16 conversion
// ---------------------------------------------------------------------------
__global__ __launch_bounds__(256) void cvt_f32_to_f16(const float* __restrict__ in,
                                                      _Float16* __restrict__ out,
                                                      int n) {
  int idx = blockIdx.x * 256 + threadIdx.x;
  if (idx < n) out[idx] = (_Float16)in[idx];
}

// ---------------------------------------------------------------------------
// WMMA GEMM:  out[m, a] = sum_k X[m,k] * W[a,k] + bias[a]
// X: (Mm x Dd) f16 row-major, W: (Aa x Dd) f16 row-major (acts as B = W^T).
// One wave per 16x16 output tile; K=512 -> 16 x v_wmma_f32_16x16x32_f16.
// Fragment layouts per CDNA5 ISA 7.12.2 (wave32):
//   A (16x32 f16): lane L -> m = L&15, kb = (L>>4)*8;
//       halfs[0..7]  = K = k0+kb+0..7      (16B contiguous)
//       halfs[8..15] = K = k0+kb+16..23    (16B contiguous)
//   B (32x16 f16): lane L -> n = L&15;
//       halfs[0..7] = K = k0+kb+0..7, halfs[8..15] = K = k0+kb+16..23, kb as A (two 16B loads)
//   C/D (16x16 f32): VGPR r, lane L -> M = r + 8*(L>>4), N = L&15
// ---------------------------------------------------------------------------
__global__ __launch_bounds__(256) void gemm_wmma_f16(const _Float16* __restrict__ X,
                                                     const _Float16* __restrict__ W,
                                                     const float* __restrict__ bias,
                                                     float* __restrict__ out) {
  const int lane = threadIdx.x & 31;
  const int wave = threadIdx.x >> 5;
  const int tile = blockIdx.x * 8 + wave;       // 8192 tiles total
  const int mtile = tile >> 5;                  // 0..255
  const int atile = tile & 31;                  // 0..31

  const int mA  = lane & 15;
  const int kbA = (lane >> 4) * 8;
  const int nB  = lane & 15;
  const int kbB = (lane >> 4) * 8;

  const _Float16* __restrict__ xrow = X + (size_t)(mtile * 16 + mA) * Dd;
  const _Float16* __restrict__ wrow = W + (size_t)(atile * 16 + nB) * Dd;

  v8f c = {};
  for (int k0 = 0; k0 < Dd; k0 += 32) {
    union { v16h v; v8h h[2]; } a, b;
    a.h[0] = *(const v8h*)(xrow + k0 + kbA);        // K = k0+kb+0..7
    a.h[1] = *(const v8h*)(xrow + k0 + kbA + 16);   // K = k0+kb+16..23
    b.h[0] = *(const v8h*)(wrow + k0 + kbB);        // K = k0+kbB+0..7
    b.h[1] = *(const v8h*)(wrow + k0 + kbB + 16);   // K = k0+kbB+16..23
    c = __builtin_amdgcn_wmma_f32_16x16x32_f16(
        /*neg_a=*/false, a.v, /*neg_b=*/false, b.v,
        /*c_mod=*/(short)0, c, /*reuse_a=*/false, /*reuse_b=*/false);
  }

  const int nOut = atile * 16 + (lane & 15);
  const float bv = bias[nOut];
  const int mhi = (lane >> 4) * 8;
  const int mBase = mtile * 16 + mhi;
#pragma unroll
  for (int r = 0; r < 8; ++r) {
    out[(size_t)(mBase + r) * Aa + nOut] = c[r] + bv;
  }
}

// ---------------------------------------------------------------------------
// Phase 2 (dominant cost, TRANS-bound):
//   out[b,i,j] = sum_a tanh(x1[b,j,a] + x2[b,i+1,a]) * Wout[a] + bout
// Grid (127, 32): block owns one (b, i). x2 row + Wout staged in LDS.
// 8 waves; wave handles one j per pass, 16 passes; lane strides a by 32
// (coalesced 128B x1 reads, L2-resident); wave32 shfl_xor reduction.
// ---------------------------------------------------------------------------
__global__ __launch_bounds__(256) void attn_tanh_reduce(const float* __restrict__ x1,
                                                        const float* __restrict__ x2,
                                                        const float* __restrict__ Wout,
                                                        const float* __restrict__ bout,
                                                        float* __restrict__ out) {
  __shared__ float x2s[Aa];
  __shared__ float wos[Aa];

  const int i = blockIdx.x;   // 0..126
  const int b = blockIdx.y;   // 0..31

  const float* __restrict__ x2row = x2 + ((size_t)b * Nn + (i + 1)) * Aa;
  for (int t = threadIdx.x; t < Aa; t += 256) {
    x2s[t] = x2row[t];
    wos[t] = Wout[t];
  }
  __syncthreads();

  const int lane = threadIdx.x & 31;
  const int wave = threadIdx.x >> 5;
  const float bo = bout[0];

  for (int jp = 0; jp < 16; ++jp) {
    const int j = jp * 8 + wave;
    const float* __restrict__ x1row = x1 + ((size_t)b * Nn + j) * Aa;
    float acc = 0.0f;
#pragma unroll 4
    for (int a = lane; a < Aa; a += 32) {
      acc += fast_tanh(x1row[a] + x2s[a]) * wos[a];
    }
#pragma unroll
    for (int off = 16; off > 0; off >>= 1) {
      acc += __shfl_xor(acc, off, 32);
    }
    if (lane == 0) {
      out[((size_t)b * 127 + i) * Nn + j] = acc + bo;
    }
  }
}

// ---------------------------------------------------------------------------
// launch
// ---------------------------------------------------------------------------
extern "C" void kernel_launch(void* const* d_in, const int* in_sizes, int n_in,
                              void* d_out, int out_size, void* d_ws, size_t ws_size,
                              hipStream_t stream) {
  const float* x    = (const float*)d_in[0];  // (32,128,512)
  const float* W1   = (const float*)d_in[1];  // (512,512)
  const float* b1   = (const float*)d_in[2];  // (512,)
  const float* W2   = (const float*)d_in[3];  // (512,512)
  const float* b2   = (const float*)d_in[4];  // (512,)
  const float* Wout = (const float*)d_in[5];  // (512,)
  const float* bout = (const float*)d_in[6];  // (1,)
  float* out = (float*)d_out;                 // (32,127,128)

  // Workspace layout (21 MB total)
  char* ws = (char*)d_ws;
  constexpr size_t xhBytes = (size_t)Mm * Dd * 2;        // 4 MB
  constexpr size_t whBytes = (size_t)Aa * Dd * 2;        // 512 KB each
  constexpr size_t hBytes  = (size_t)Mm * Aa * 4;        // 8 MB each
  _Float16* xh  = (_Float16*)(ws);
  _Float16* W1h = (_Float16*)(ws + xhBytes);
  _Float16* W2h = (_Float16*)(ws + xhBytes + whBytes);
  float*    x1  = (float*)   (ws + xhBytes + 2 * whBytes);
  float*    x2  = (float*)   (ws + xhBytes + 2 * whBytes + hBytes);

  // Phase 0: f32 -> f16 conversions
  {
    int nx = Mm * Dd;          // 2,097,152
    int nw = Aa * Dd;          // 262,144
    cvt_f32_to_f16<<<(nx + 255) / 256, 256, 0, stream>>>(x, xh, nx);
    cvt_f32_to_f16<<<(nw + 255) / 256, 256, 0, stream>>>(W1, W1h, nw);
    cvt_f32_to_f16<<<(nw + 255) / 256, 256, 0, stream>>>(W2, W2h, nw);
  }

  // Phase 1: WMMA GEMMs (8192 tiles, 8 waves/block -> 1024 blocks)
  gemm_wmma_f16<<<1024, 256, 0, stream>>>(xh, W1h, b1, x1);
  gemm_wmma_f16<<<1024, 256, 0, stream>>>(xh, W2h, b2, x2);

  // Phase 2: tanh + weighted reduction (dominant, TRANS-bound)
  attn_tanh_reduce<<<dim3(127, 32), 256, 0, stream>>>(x1, x2, Wout, bout, out);
}
